// GTLayer_3599182594505
// MI455X (gfx1250) — matrix operand order repaired and checked by reading, not compiled, not run on hardware
//
#include <hip/hip_runtime.h>
#include <hip/hip_bf16.h>

typedef __attribute__((ext_vector_type(16))) __bf16 v16bf;
typedef __attribute__((ext_vector_type(8)))  float  v8f;

#define EMB 128
#define HEADS 4
// wfrag layout: [weight(3)][colTile(8)][kChunk(4)][lane(32)][elem(16)] bf16
#define FRAG_ELEMS 16
#define FRAGS_PER_WEIGHT (8 * 4 * 32)        // 1024 fragments
#define WFRAG_TOTAL (3 * FRAGS_PER_WEIGHT)   // 3072 fragments

// ---------------------------------------------------------------------------
// Zero-fill (harness poisons d_out / d_ws with 0xAA before timing)
// ---------------------------------------------------------------------------
__global__ void gt_zero_f32(float* __restrict__ p, int n) {
    int i = blockIdx.x * blockDim.x + threadIdx.x;
    if (i < n) p[i] = 0.0f;
}

// ---------------------------------------------------------------------------
// Pre-swizzle the three 128x128 f32 weight matrices into bf16 B-matrix
// fragments for v_wmma_f32_16x16x32_bf16 in contiguous k order (same as A):
//   lane < 16 : column N = lane,      elem e -> K = kc*32 + e       (K 0..15)
//   lane >=16 : column N = lane - 16, elem e -> K = kc*32 + 16 + e  (K 16..31)
// i.e. K = kc*32 + (lane>>4)*16 + e
// ---------------------------------------------------------------------------
__global__ void gt_prep_wfrag(const float* __restrict__ qW,
                              const float* __restrict__ kW,
                              const float* __restrict__ vW,
                              __bf16* __restrict__ wfrag) {
    int idx = blockIdx.x * blockDim.x + threadIdx.x;
    if (idx >= WFRAG_TOTAL) return;
    int lane = idx & 31;
    int kc   = (idx >> 5) & 3;
    int tile = (idx >> 7) & 7;
    int w    = idx >> 10;
    const float* W = (w == 0) ? qW : (w == 1) ? kW : vW;
    int half = lane >> 4;
    int n    = lane & 15;
    __bf16* out = wfrag + idx * FRAG_ELEMS;
#pragma unroll
    for (int e = 0; e < FRAG_ELEMS; ++e) {
        int K = kc * 32 + half * 16 + e;
        out[e] = (__bf16)W[K * EMB + tile * 16 + n];
    }
}

// ---------------------------------------------------------------------------
// Node-level QKV projection: Q = embeds @ qW, K = embeds @ kW, V = embeds @ vW
// One block = 16 nodes x 128 cols x {Q,K,V}. 8 waves; wave w owns col tile w.
// 12 WMMAs per wave (4 k-chunks x 3 weights), f32 accumulate.
// N_NODES = 50000 is an exact multiple of 16 -> no tail handling.
// ---------------------------------------------------------------------------
__global__ __launch_bounds__(256) void gt_qkv_gemm(
        const float*  __restrict__ embeds,
        const __bf16* __restrict__ wfrag,
        float* __restrict__ Q, float* __restrict__ K, float* __restrict__ V) {
    const int wave = threadIdx.x >> 5;
    const int lane = threadIdx.x & 31;
    const int half = lane >> 4;
    const int lm   = lane & 15;
    const int nodeBase = blockIdx.x * 16;

    // A-fragment source row: lane holds row M = lm for both halves.
    const float* arow = embeds + (size_t)(nodeBase + lm) * EMB;

    v8f accQ = {}, accK = {}, accV = {};
    const int fragStrideW = FRAGS_PER_WEIGHT * FRAG_ELEMS;   // per-weight stride

#pragma unroll
    for (int kc = 0; kc < 4; ++kc) {
        // Build bf16 A fragment in contiguous k order (same as wfrag B):
        //   VGPR pair p, half h: K = kc*32 + 16*h + 2p + {0,1}
        v16bf a;
#pragma unroll
        for (int p = 0; p < 8; ++p) {
            int k0 = kc * 32 + half * 16 + 2 * p;
            float2 f = *(const float2*)(arow + k0);
            a[2 * p]     = (__bf16)f.x;
            a[2 * p + 1] = (__bf16)f.y;
        }
        const __bf16* bq = wfrag + (size_t)(((wave * 4 + kc) * 32 + lane)) * FRAG_ELEMS;
        v16bf bQ = *(const v16bf*)(bq);
        v16bf bK = *(const v16bf*)(bq + fragStrideW);
        v16bf bV = *(const v16bf*)(bq + 2 * fragStrideW);

        accQ = __builtin_amdgcn_wmma_f32_16x16x32_bf16(false, a, false, bQ,
                                                       (short)0, accQ, false, false);
        accK = __builtin_amdgcn_wmma_f32_16x16x32_bf16(false, a, false, bK,
                                                       (short)0, accK, false, false);
        accV = __builtin_amdgcn_wmma_f32_16x16x32_bf16(false, a, false, bV,
                                                       (short)0, accV, false, false);
    }

    // C/D layout: elem r -> row M = r + 8*half, col N = lm (within 16-wide tile)
    const int col = wave * 16 + lm;
#pragma unroll
    for (int r = 0; r < 8; ++r) {
        int node = nodeBase + r + half * 8;
        size_t o = (size_t)node * EMB + col;
        Q[o] = accQ[r];
        K[o] = accK[r];
        V[o] = accV[r];
    }
}

// ---------------------------------------------------------------------------
// Pass 1: per-edge per-head logits. Warp per edge.
// Lane loads 4 consecutive dims (float4); head h owns lanes 8h..8h+7.
// 8-lane shuffle reduction -> clip -> exp -> store + atomic segment sum.
// ---------------------------------------------------------------------------
__global__ __launch_bounds__(256) void gt_edge_logits(
        const float* __restrict__ Q, const float* __restrict__ Kf,
        const int* __restrict__ rows, const int* __restrict__ cols,
        float* __restrict__ expAtt, float* __restrict__ attNorm, int n_edges) {
    int e    = (blockIdx.x * blockDim.x + threadIdx.x) >> 5;
    int lane = threadIdx.x & 31;
    if (e >= n_edges) return;
    int r = rows[e];
    int c = cols[e];
    float4 q = *(const float4*)(Q  + (size_t)r * EMB + lane * 4);
    float4 k = *(const float4*)(Kf + (size_t)c * EMB + lane * 4);
    float part = q.x * k.x + q.y * k.y + q.z * k.z + q.w * k.w;
    part += __shfl_down(part, 4, 32);
    part += __shfl_down(part, 2, 32);
    part += __shfl_down(part, 1, 32);
    if ((lane & 7) == 0) {
        int h = lane >> 3;
        float att = fminf(fmaxf(part, -10.0f), 10.0f);
        float ex  = __expf(att);
        expAtt[(size_t)e * HEADS + h] = ex;
        atomicAdd(&attNorm[(size_t)r * HEADS + h], ex);
    }
}

// ---------------------------------------------------------------------------
// Pass 2: out[row] += (expAtt / (norm[row]+1e-8)) * V[col]. Warp per edge.
// Lane's 4 dims all belong to head = lane>>3. Atomics land in L2 (out = 25.6MB).
// ---------------------------------------------------------------------------
__global__ __launch_bounds__(256) void gt_edge_aggregate(
        const float* __restrict__ V,
        const int* __restrict__ rows, const int* __restrict__ cols,
        const float* __restrict__ expAtt, const float* __restrict__ attNorm,
        float* __restrict__ out, int n_edges) {
    int e    = (blockIdx.x * blockDim.x + threadIdx.x) >> 5;
    int lane = threadIdx.x & 31;
    if (e >= n_edges) return;
    int r = rows[e];
    int c = cols[e];
    int h = lane >> 3;
    float scale = expAtt[(size_t)e * HEADS + h] /
                  (attNorm[(size_t)r * HEADS + h] + 1e-8f);
    float4 v = *(const float4*)(V + (size_t)c * EMB + lane * 4);
    float* o = out + (size_t)r * EMB + lane * 4;
    atomicAdd(o + 0, scale * v.x);
    atomicAdd(o + 1, scale * v.y);
    atomicAdd(o + 2, scale * v.z);
    atomicAdd(o + 3, scale * v.w);
}

// ---------------------------------------------------------------------------
// Launch
// ---------------------------------------------------------------------------
extern "C" void kernel_launch(void* const* d_in, const int* in_sizes, int n_in,
                              void* d_out, int out_size, void* d_ws, size_t ws_size,
                              hipStream_t stream) {
    const float* embeds = (const float*)d_in[0];
    const float* qW     = (const float*)d_in[1];
    const float* kW     = (const float*)d_in[2];
    const float* vW     = (const float*)d_in[3];
    const int*   rows   = (const int*)d_in[4];
    const int*   cols   = (const int*)d_in[5];

    const int N = in_sizes[0] / EMB;   // 50000
    const int E = in_sizes[4];         // 600000
    float* out = (float*)d_out;

    // Workspace carve-up (all offsets multiples of 256 bytes)
    char* ws = (char*)d_ws;
    size_t qkvBytes = (size_t)N * EMB * sizeof(float);      // 25.6 MB each
    float*  Q       = (float*)(ws);
    float*  K       = (float*)(ws + qkvBytes);
    float*  V       = (float*)(ws + 2 * qkvBytes);
    float*  expAtt  = (float*)(ws + 3 * qkvBytes);                            // E*4 f32
    float*  attNorm = (float*)(ws + 3 * qkvBytes + (size_t)E * HEADS * 4);    // N*4 f32
    __bf16* wfrag   = (__bf16*)(ws + 3 * qkvBytes + (size_t)E * HEADS * 4
                                                 + (size_t)N * HEADS * 4);

    // 1. Weight fragments (bf16, WMMA B layout)
    gt_prep_wfrag<<<(WFRAG_TOTAL + 255) / 256, 256, 0, stream>>>(qW, kW, vW, wfrag);

    // 2. Zero the accumulation buffers
    gt_zero_f32<<<(out_size + 255) / 256, 256, 0, stream>>>(out, out_size);
    gt_zero_f32<<<(N * HEADS + 255) / 256, 256, 0, stream>>>(attNorm, N * HEADS);

    // 3. Node-level QKV projection with WMMA (N multiple of 16)
    gt_qkv_gemm<<<N / 16, 256, 0, stream>>>(embeds, wfrag, Q, K, V);

    // 4. Edge logits + segment softmax denominator
    {
        long long threads = (long long)E * 32;
        int blocks = (int)((threads + 255) / 256);
        gt_edge_logits<<<blocks, 256, 0, stream>>>(Q, K, rows, cols, expAtt, attNorm, E);
    }

    // 5. Weighted segment sum of V
    {
        long long threads = (long long)E * 32;
        int blocks = (int)((threads + 255) / 256);
        gt_edge_aggregate<<<blocks, 256, 0, stream>>>(V, rows, cols, expAtt, attNorm, out, E);
    }
}
